// PoseMFShapeGaussianNet_21449066676759
// MI455X (gfx1250) — matrix operand order repaired and checked by reading, not compiled, not run on hardware
//
#include <hip/hip_runtime.h>

// ---------------------------------------------------------------------------
// PoseMF/Shape Gaussian net forward for MI455X (gfx1250, wave32, WMMA).
// GEMMs: v_wmma_f32_16x16x32_bf16, double-buffered LDS pipeline, A-tile via
// GLOBAL_LOAD_ASYNC_TO_LDS_B128 (ASYNCcnt) when the builtin is available.
// ---------------------------------------------------------------------------

typedef __bf16 bf16_t;
typedef __bf16        v16bf __attribute__((ext_vector_type(16)));
typedef float         v8f   __attribute__((ext_vector_type(8)));
typedef unsigned int  v8u   __attribute__((ext_vector_type(8)));

#define BROWS 4096
#define NJ    23

#if defined(__HIP_DEVICE_COMPILE__) && __has_builtin(__builtin_amdgcn_global_load_async_to_lds_b128)
#define USE_ASYNC_LDS 1
#else
#define USE_ASYNC_LDS 0
#endif

#if USE_ASYNC_LDS
// Builtin prototype (from clang diagnostic): param0 = v4i* in global AS (1),
// param1 = v4i* in LDS AS (3), then two immediate ints (offset, cpol).
typedef int v4i_t __attribute__((vector_size(16)));
typedef __attribute__((address_space(1))) v4i_t* as1_v4i;
typedef __attribute__((address_space(3))) v4i_t* as3_v4i;

__device__ __forceinline__ void async_cp16(const void* g, void* l)
{
    // flat->AS1 is bit-identical; flat LDS addr[31:0] is the LDS offset (AS3 is 32-bit)
    __builtin_amdgcn_global_load_async_to_lds_b128(
        (as1_v4i)(unsigned long long)g,
        (as3_v4i)(unsigned int)(unsigned long long)l, 0, 0);
}
__device__ __forceinline__ void wait_async()
{
#if __has_builtin(__builtin_amdgcn_s_wait_asynccnt)
    __builtin_amdgcn_s_wait_asynccnt(0);
#else
    asm volatile("s_wait_asynccnt 0x0" ::: "memory");
#endif
}
#endif

// ======================= WMMA GEMM =========================================
// C[M,N] = act(A[M,K] * W[K,N] + bias); A/W bf16 row-major; K%32==0,
// M%128==0, N%64==0. 256 thr = 8 waves; block tile 128x64; each wave owns a
// 32x32 tile as 2x2 16x16 fragments. Double-buffered LDS, 1 barrier/K-step.
__global__ __launch_bounds__(256)
void gemm_bf16_wmma(const bf16_t* __restrict__ A, const bf16_t* __restrict__ W,
                    const float* __restrict__ bias, float* __restrict__ outF,
                    bf16_t* __restrict__ outB, int M, int N, int K, int act)
{
    __shared__ unsigned int   sA[2][128][20];  // 32 bf16/row (16 dw) + pad; 16B-aligned rows
    __shared__ unsigned short sB[2][64][34];   // N-major transposed B tile + pad

    const int tid  = threadIdx.x;
    const int lane = tid & 31;
    const int wave = tid >> 5;
    const int wm   = (wave & 3) * 32;
    const int wn   = (wave >> 2) * 32;
    const int bm   = blockIdx.y * 128;
    const int bn   = blockIdx.x * 64;
    const int hi   = (lane >= 16) ? 1 : 0;
    const int l15  = lane & 15;

    // per-thread copy coordinates: A = 512 uint4 (2/thread), B = 256 uint4 (1/thread)
    const int ar0 = tid >> 2,          as0 = tid & 3;
    const int ar1 = (tid + 256) >> 2,  as1 = (tid + 256) & 3;
    const int bkr = tid >> 3,          bns = tid & 7;

    v8f z = {0.f,0.f,0.f,0.f,0.f,0.f,0.f,0.f};
    v8f acc00 = z, acc01 = z, acc10 = z, acc11 = z;
    const int ab = hi ? 8 : 0;   // A-frag dword col base (K-half select: lane half h holds k 16h..16h+15)
    const int bb = hi ? 8 : 0;   // B-frag dword col base
    const int nKt = K >> 5;

    // ---- prologue: stage K-tile 0 into buffer 0 ----
    {
#if USE_ASYNC_LDS
        async_cp16(A + (size_t)(bm + ar0) * K + as0 * 8, &sA[0][ar0][as0 * 4]);
        async_cp16(A + (size_t)(bm + ar1) * K + as1 * 8, &sA[0][ar1][as1 * 4]);
#else
        uint4 va0 = *(const uint4*)(A + (size_t)(bm + ar0) * K + as0 * 8);
        uint4 va1 = *(const uint4*)(A + (size_t)(bm + ar1) * K + as1 * 8);
        *(uint4*)&sA[0][ar0][as0 * 4] = va0;
        *(uint4*)&sA[0][ar1][as1 * 4] = va1;
#endif
        uint4 vb = *(const uint4*)(W + (size_t)bkr * N + bn + bns * 8);
        const unsigned short* ps = (const unsigned short*)&vb;
        #pragma unroll
        for (int i = 0; i < 8; ++i) sB[0][bns * 8 + i][bkr] = ps[i];
#if USE_ASYNC_LDS
        wait_async();
#endif
        __syncthreads();
    }

    for (int kt = 0; kt < nKt; ++kt) {
        const int cur = kt & 1, nxt = cur ^ 1;
        const bool hasNext = (kt + 1) < nKt;
        uint4 vb{};
#if !USE_ASYNC_LDS
        uint4 va0{}, va1{};
#endif
        if (hasNext) {           // issue next tile's loads before computing
            const int kn = (kt + 1) << 5;
#if USE_ASYNC_LDS
            async_cp16(A + (size_t)(bm + ar0) * K + kn + as0 * 8, &sA[nxt][ar0][as0 * 4]);
            async_cp16(A + (size_t)(bm + ar1) * K + kn + as1 * 8, &sA[nxt][ar1][as1 * 4]);
#else
            va0 = *(const uint4*)(A + (size_t)(bm + ar0) * K + kn + as0 * 8);
            va1 = *(const uint4*)(A + (size_t)(bm + ar1) * K + kn + as1 * 8);
#endif
            vb = *(const uint4*)(W + (size_t)(kn + bkr) * N + bn + bns * 8);
        }
        if (kt + 2 < nKt) {      // gfx1250 global_prefetch_b8, distance 2
            const int kf = (kt + 2) << 5;
            __builtin_prefetch(A + (size_t)(bm + (tid >> 1)) * K + kf, 0, 1);
            __builtin_prefetch(W + (size_t)(kf + (tid >> 3)) * N + bn, 0, 1);
        }

        // ---- fragments from current buffer (contiguous k = 16h + s for both A and B) ----
        v8u ta, tb2;
        const unsigned int* arp0 = &sA[cur][wm + l15][0];
        const unsigned int* arp1 = &sA[cur][wm + 16 + l15][0];
        #pragma unroll
        for (int j = 0; j < 8; ++j) ta[j] = arp0[ab + j];
        v16bf a0 = __builtin_bit_cast(v16bf, ta);
        #pragma unroll
        for (int j = 0; j < 8; ++j) ta[j] = arp1[ab + j];
        v16bf a1 = __builtin_bit_cast(v16bf, ta);
        const unsigned int* brp0 = (const unsigned int*)&sB[cur][wn + l15][0];
        const unsigned int* brp1 = (const unsigned int*)&sB[cur][wn + 16 + l15][0];
        #pragma unroll
        for (int j = 0; j < 8; ++j) tb2[j] = brp0[bb + j];
        v16bf b0 = __builtin_bit_cast(v16bf, tb2);
        #pragma unroll
        for (int j = 0; j < 8; ++j) tb2[j] = brp1[bb + j];
        v16bf b1 = __builtin_bit_cast(v16bf, tb2);

        acc00 = __builtin_amdgcn_wmma_f32_16x16x32_bf16(false, a0, false, b0, (short)0, acc00, false, false);
        acc01 = __builtin_amdgcn_wmma_f32_16x16x32_bf16(false, a0, false, b1, (short)0, acc01, false, false);
        acc10 = __builtin_amdgcn_wmma_f32_16x16x32_bf16(false, a1, false, b0, (short)0, acc10, false, false);
        acc11 = __builtin_amdgcn_wmma_f32_16x16x32_bf16(false, a1, false, b1, (short)0, acc11, false, false);

        if (hasNext) {           // commit next tile into the other buffer
#if !USE_ASYNC_LDS
            *(uint4*)&sA[nxt][ar0][as0 * 4] = va0;
            *(uint4*)&sA[nxt][ar1][as1 * 4] = va1;
#endif
            const unsigned short* ps = (const unsigned short*)&vb;
            #pragma unroll
            for (int i = 0; i < 8; ++i) sB[nxt][bns * 8 + i][bkr] = ps[i];
#if USE_ASYNC_LDS
            wait_async();
#endif
        }
        __syncthreads();
    }

    // ---- epilogue: bias + ELU, f32 and/or bf16 stores ----
    #pragma unroll
    for (int fm = 0; fm < 2; ++fm) {
        #pragma unroll
        for (int fn = 0; fn < 2; ++fn) {
            v8f accv = (fm == 0) ? ((fn == 0) ? acc00 : acc01)
                                 : ((fn == 0) ? acc10 : acc11);
            int col = bn + wn + fn * 16 + l15;
            float bv = bias ? bias[col] : 0.f;
            #pragma unroll
            for (int r = 0; r < 8; ++r) {
                int row = bm + wm + fm * 16 + r + hi * 8;
                float v = accv[r] + bv;
                if (act) v = (v > 0.f) ? v : (__expf(v) - 1.f);
                size_t o = (size_t)row * N + col;
                if (outF) outF[o] = v;
                if (outB) outB[o] = (bf16_t)v;
            }
        }
    }
}

// ======================= packing kernels ===================================
__global__ void pack_w_bf16(const float* __restrict__ W, bf16_t* __restrict__ out,
                            int K, int N, int Kpad)
{
    size_t idx = (size_t)blockIdx.x * 256 + threadIdx.x;
    size_t tot = (size_t)Kpad * N;
    if (idx >= tot) return;
    int k = (int)(idx / N), n = (int)(idx % N);
    out[idx] = (k < K) ? (bf16_t)W[(size_t)k * N + n] : (bf16_t)0.f;
}

__global__ void pack_embed_kernel(const float* __restrict__ xin, const float* __restrict__ sp,
                                  const float* __restrict__ glob, const float* __restrict__ cam,
                                  bf16_t* __restrict__ out)
{
    size_t idx = (size_t)blockIdx.x * 256 + threadIdx.x;
    const size_t tot = (size_t)BROWS * 2080;
    if (idx >= tot) return;
    int b = (int)(idx / 2080), c = (int)(idx % 2080);
    float v;
    if      (c < 2048) v = xin[(size_t)b * 2048 + c];
    else if (c < 2068) v = sp[b * 20 + (c - 2048)];
    else if (c < 2074) v = glob[b * 6 + (c - 2068)];
    else if (c < 2077) v = cam[b * 3 + (c - 2074)];
    else               v = 0.f;
    out[idx] = (bf16_t)v;
}

struct Par8 { int p[8]; };

__global__ void pack_feat_kernel(const bf16_t* __restrict__ embed,
                                 const float* __restrict__ UpAll,
                                 const float* __restrict__ SpAll,
                                 const float* __restrict__ RAll,
                                 Par8 par, int np, int Kpad, bf16_t* __restrict__ out)
{
    size_t idx = (size_t)blockIdx.x * 256 + threadIdx.x;
    size_t tot = (size_t)BROWS * Kpad;
    if (idx >= tot) return;
    int b = (int)(idx / Kpad), c = (int)(idx % Kpad);
    if (c < 512) { out[idx] = embed[(size_t)b * 512 + c]; return; }
    int d = c - 512;
    int n9 = 9 * np, n12 = 12 * np, n21 = 21 * np;
    float v = 0.f;
    if (d < n9)       { int i = d / 9, e = d % 9;
                        v = UpAll[((size_t)b * NJ + par.p[i]) * 9 + e]; }
    else if (d < n12) { int d2 = d - n9; int i = d2 / 3, e = d2 % 3;
                        v = SpAll[((size_t)b * NJ + par.p[i]) * 3 + e]; }
    else if (d < n21) { int d3 = d - n12; int i = d3 / 9, e = d3 % 9;
                        v = RAll[((size_t)b * NJ + par.p[i]) * 9 + e]; }
    out[idx] = (bf16_t)v;
}

// ======================= small heads =======================================
__global__ __launch_bounds__(128)
void heads_kernel(const float* __restrict__ x,
                  const float* __restrict__ Wsh, const float* __restrict__ bsh,
                  const float* __restrict__ Wg,  const float* __restrict__ bg,
                  const float* __restrict__ Wc,  const float* __restrict__ bc,
                  float* __restrict__ shape_params, float* __restrict__ out_mean,
                  float* __restrict__ out_std, float* __restrict__ out_glob,
                  float* __restrict__ out_cam)
{
    __shared__ float xs[1024];
    int row = blockIdx.x;
    for (int i = threadIdx.x; i < 1024; i += 128) xs[i] = x[(size_t)row * 1024 + i];
    __syncthreads();
    int o = threadIdx.x;
    if (o >= 29) return;
    float s = 0.f;
    if (o < 20) {
        for (int k = 0; k < 1024; ++k) s += xs[k] * Wsh[k * 20 + o];
        s += bsh[o];
        shape_params[row * 20 + o] = s;
        if (o < 10) out_mean[row * 10 + o] = s;
        else        out_std[row * 10 + (o - 10)] = __expf(s);
    } else if (o < 26) {
        int c = o - 20;
        for (int k = 0; k < 1024; ++k) s += xs[k] * Wg[k * 6 + c];
        s += bg[c] + ((c == 0 || c == 3) ? 1.f : 0.f);
        out_glob[row * 6 + c] = s;
    } else {
        int c = o - 26;
        for (int k = 0; k < 1024; ++k) s += xs[k] * Wc[k * 3 + c];
        s += bc[c] + ((c == 0) ? 0.9f : 0.f);
        out_cam[row * 3 + c] = s;
    }
}

// ======================= l2 + 3x3 SVD ======================================
template<int P, int Q>
__device__ __forceinline__ void jacobi3(float A[3][3], float V[3][3])
{
    float apq = A[P][Q];
    if (fabsf(apq) < 1e-20f) return;
    float theta = (A[Q][Q] - A[P][P]) / (2.f * apq);
    float t = copysignf(1.f, theta) / (fabsf(theta) + sqrtf(theta * theta + 1.f));
    float c = 1.f / sqrtf(t * t + 1.f);
    float s = t * c;
    #pragma unroll
    for (int k = 0; k < 3; ++k) { float a0 = A[k][P], a1 = A[k][Q];
        A[k][P] = c * a0 - s * a1; A[k][Q] = s * a0 + c * a1; }
    #pragma unroll
    for (int k = 0; k < 3; ++k) { float a0 = A[P][k], a1 = A[Q][k];
        A[P][k] = c * a0 - s * a1; A[Q][k] = s * a0 + c * a1; }
    #pragma unroll
    for (int k = 0; k < 3; ++k) { float v0 = V[k][P], v1 = V[k][Q];
        V[k][P] = c * v0 - s * v1; V[k][Q] = s * v0 + c * v1; }
}

__global__ __launch_bounds__(128)
void l2_svd_kernel(const float* __restrict__ h, const float* __restrict__ W2,
                   const float* __restrict__ b2, int j,
                   float* __restrict__ outF, float* __restrict__ outU,
                   float* __restrict__ outS, float* __restrict__ outV,
                   float* __restrict__ outR,
                   float* __restrict__ UpAll, float* __restrict__ SpAll)
{
    __shared__ float w2s[2304];
    __shared__ float b2s[9];
    __shared__ float ht[128][33];
    for (int i = threadIdx.x; i < 2304; i += 128) w2s[i] = W2[i];
    if (threadIdx.x < 9) b2s[threadIdx.x] = b2[threadIdx.x];

    int rb = blockIdx.x * 128;
    float f[9];
    #pragma unroll
    for (int i = 0; i < 9; ++i) f[i] = 0.f;

    for (int kc = 0; kc < 256; kc += 32) {
        __syncthreads();
        for (int i = threadIdx.x; i < 128 * 32; i += 128) {
            int r = i >> 5, k = i & 31;
            ht[r][k] = h[(size_t)(rb + r) * 256 + kc + k];
        }
        __syncthreads();
        for (int k = 0; k < 32; ++k) {
            float hv = ht[threadIdx.x][k];
            const float* wr = &w2s[(kc + k) * 9];
            #pragma unroll
            for (int i = 0; i < 9; ++i) f[i] += hv * wr[i];
        }
    }

    int row = rb + threadIdx.x;
    #pragma unroll
    for (int i = 0; i < 9; ++i) f[i] += b2s[i];
    f[0] += 5.f; f[4] += 5.f; f[8] += 5.f;   // F = lin + 5*I

    float Fm[3][3] = {{f[0], f[1], f[2]}, {f[3], f[4], f[5]}, {f[6], f[7], f[8]}};
    float Am[3][3], Vm[3][3] = {{1,0,0},{0,1,0},{0,0,1}};
    #pragma unroll
    for (int r = 0; r < 3; ++r)
        #pragma unroll
        for (int c = 0; c < 3; ++c)
            Am[r][c] = Fm[0][r]*Fm[0][c] + Fm[1][r]*Fm[1][c] + Fm[2][r]*Fm[2][c];

    #pragma unroll
    for (int sweep = 0; sweep < 6; ++sweep) {
        jacobi3<0,1>(Am, Vm); jacobi3<0,2>(Am, Vm); jacobi3<1,2>(Am, Vm);
    }

    float d0 = Am[0][0], d1 = Am[1][1], d2 = Am[2][2];
#define SORT_SWAP(da, db, ca, cb)                                              \
    if (da < db) { float td = da; da = db; db = td;                            \
        float t0 = Vm[0][ca]; Vm[0][ca] = Vm[0][cb]; Vm[0][cb] = t0;           \
        float t1 = Vm[1][ca]; Vm[1][ca] = Vm[1][cb]; Vm[1][cb] = t1;           \
        float t2 = Vm[2][ca]; Vm[2][ca] = Vm[2][cb]; Vm[2][cb] = t2; }
    SORT_SWAP(d0, d1, 0, 1)
    SORT_SWAP(d1, d2, 1, 2)
    SORT_SWAP(d0, d1, 0, 1)
#undef SORT_SWAP

    float Um[3][3], Sv[3];
    #pragma unroll
    for (int i = 0; i < 3; ++i) {
        float u0 = Fm[0][0]*Vm[0][i] + Fm[0][1]*Vm[1][i] + Fm[0][2]*Vm[2][i];
        float u1 = Fm[1][0]*Vm[0][i] + Fm[1][1]*Vm[1][i] + Fm[1][2]*Vm[2][i];
        float u2 = Fm[2][0]*Vm[0][i] + Fm[2][1]*Vm[1][i] + Fm[2][2]*Vm[2][i];
        float nrm = sqrtf(u0*u0 + u1*u1 + u2*u2);
        Sv[i] = nrm;
        float inv = (nrm > 1e-20f) ? 1.f / nrm : 0.f;
        Um[0][i] = u0 * inv; Um[1][i] = u1 * inv; Um[2][i] = u2 * inv;
    }

    float detU = Um[0][0]*(Um[1][1]*Um[2][2]-Um[1][2]*Um[2][1])
               - Um[0][1]*(Um[1][0]*Um[2][2]-Um[1][2]*Um[2][0])
               + Um[0][2]*(Um[1][0]*Um[2][1]-Um[1][1]*Um[2][0]);
    float detV = Vm[0][0]*(Vm[1][1]*Vm[2][2]-Vm[1][2]*Vm[2][1])
               - Vm[0][1]*(Vm[1][0]*Vm[2][2]-Vm[1][2]*Vm[2][0])
               + Vm[0][2]*(Vm[1][0]*Vm[2][1]-Vm[1][1]*Vm[2][0]);
    float su = copysignf(1.f, detU), sv = copysignf(1.f, detV);

    float Up[3][3], Vp[3][3];
    #pragma unroll
    for (int r = 0; r < 3; ++r) {
        Up[r][0] = Um[r][0]; Up[r][1] = Um[r][1]; Up[r][2] = Um[r][2] * su;
        Vp[r][0] = Vm[r][0]; Vp[r][1] = Vm[r][1]; Vp[r][2] = Vm[r][2] * sv;
    }

    size_t b9 = ((size_t)row * NJ + j) * 9;
    size_t b3 = ((size_t)row * NJ + j) * 3;
    #pragma unroll
    for (int r = 0; r < 3; ++r)
        #pragma unroll
        for (int c = 0; c < 3; ++c) {
            outF[b9 + r*3 + c] = Fm[r][c];
            outU[b9 + r*3 + c] = Um[r][c];
            outV[b9 + r*3 + c] = Vm[r][c];
            float rr = Up[r][0]*Vp[c][0] + Up[r][1]*Vp[c][1] + Up[r][2]*Vp[c][2];
            outR[b9 + r*3 + c] = rr;
            UpAll[b9 + r*3 + c] = Up[r][c];
        }
    outS[b3+0] = Sv[0]; outS[b3+1] = Sv[1]; outS[b3+2] = Sv[2];
    SpAll[b3+0] = Sv[0]; SpAll[b3+1] = Sv[1]; SpAll[b3+2] = Sv[2] * su * sv;
}

// ======================= host launcher =====================================
static const int kNP[NJ] = {0,0,0,1,1,1,2,2,2,3,3,3,3,3,4,4,4,5,5,6,6,7,7};
static const int kP[NJ][8] = {
    {0},{0},{0},{0},{1},{2},{3,0},{4,1},{5,2},{6,3,0},{7,4,1},{8,5,2},{8,5,2},
    {8,5,2},{11,8,5,2},{12,8,5,2},{13,8,5,2},{15,12,8,5,2},{16,13,8,5,2},
    {17,15,12,8,5,2},{18,16,13,8,5,2},{19,17,15,12,8,5,2},{20,18,16,13,8,5,2}};

extern "C" void kernel_launch(void* const* d_in, const int* in_sizes, int n_in,
                              void* d_out, int out_size, void* d_ws, size_t ws_size,
                              hipStream_t stream)
{
    (void)out_size; (void)ws_size;
    // --- map inputs by unique element counts (robust to pytree key order) ---
    const float *in_feats = nullptr, *fc1W = nullptr, *fc1b = nullptr,
                *shW = nullptr, *shb = nullptr, *gW = nullptr, *gb = nullptr,
                *cW = nullptr, *cb = nullptr, *eW = nullptr, *eb = nullptr;
    const float *l1W[NJ] = {}, *l1b[NJ] = {}, *l2W[NJ] = {}, *l2b[NJ] = {};
    int c1w = 0, c1b = 0, c2w = 0, c2b = 0;
    for (int i = 0; i < n_in; ++i) {
        const float* p = (const float*)d_in[i];
        switch (in_sizes[i]) {
            case 4096*2048: in_feats = p; break;
            case 2048*1024: fc1W = p; break;
            case 1024:      fc1b = p; break;
            case 1024*20:   shW = p; break;
            case 20:        shb = p; break;
            case 1024*6:    gW = p; break;
            case 6:         gb = p; break;
            case 1024*3:    cW = p; break;
            case 3:         cb = p; break;
            case 2077*512:  eW = p; break;
            case 512:       eb = p; break;
            case 256:       if (c1b < NJ) l1b[c1b++] = p; break;
            case 2304:      if (c2w < NJ) l2W[c2w++] = p; break;
            case 9:         if (c2b < NJ) l2b[c2b++] = p; break;
            default:        if (c1w < NJ) l1W[c1w++] = p; break;  // pose l1.W
        }
    }

    // --- output offsets (flat, return order) ---
    float* out = (float*)d_out;
    float* outF   = out;
    float* outU   = outF + (size_t)BROWS * NJ * 9;
    float* outS   = outU + (size_t)BROWS * NJ * 9;
    float* outV   = outS + (size_t)BROWS * NJ * 3;
    float* outR   = outV + (size_t)BROWS * NJ * 9;
    float* outMu  = outR + (size_t)BROWS * NJ * 9;
    float* outStd = outMu + (size_t)BROWS * 10;
    float* outGl  = outStd + (size_t)BROWS * 10;
    float* outCam = outGl + (size_t)BROWS * 6;

    // --- workspace bump allocator ---
    char* wsp = (char*)d_ws;
    size_t off = 0;
    auto balloc = [&](size_t bytes) -> void* {
        void* p = wsp + off; off += (bytes + 255) & ~(size_t)255; return p; };

    bf16_t* inA  = (bf16_t*)balloc((size_t)BROWS * 2048 * 2);
    bf16_t* w1fc = (bf16_t*)balloc((size_t)2048 * 1024 * 2);
    float*  x    = (float*) balloc((size_t)BROWS * 1024 * 4);
    float*  shp  = (float*) balloc((size_t)BROWS * 20 * 4);
    bf16_t* embA = (bf16_t*)balloc((size_t)BROWS * 2080 * 2);
    bf16_t* wEmb = (bf16_t*)balloc((size_t)2080 * 512 * 2);
    bf16_t* emb  = (bf16_t*)balloc((size_t)BROWS * 512 * 2);
    bf16_t* feat = (bf16_t*)balloc((size_t)BROWS * 672 * 2);
    float*  hbuf = (float*) balloc((size_t)BROWS * 256 * 4);
    float*  UpA  = (float*) balloc((size_t)BROWS * NJ * 9 * 4);
    float*  SpA  = (float*) balloc((size_t)BROWS * NJ * 3 * 4);
    bf16_t* w1p[NJ];
    int kpad[NJ], indim[NJ];
    for (int j = 0; j < NJ; ++j) {
        indim[j] = 512 + 21 * kNP[j];
        kpad[j]  = (indim[j] + 31) & ~31;
        w1p[j]   = (bf16_t*)balloc((size_t)kpad[j] * 256 * 2);
    }

    auto gblk = [](size_t tot) { return (unsigned)((tot + 255) / 256); };

    // --- fc1: x = elu(input @ W1 + b1) ---
    pack_w_bf16<<<gblk((size_t)BROWS*2048), 256, 0, stream>>>(in_feats, inA, BROWS, 2048, BROWS);
    pack_w_bf16<<<gblk((size_t)2048*1024), 256, 0, stream>>>(fc1W, w1fc, 2048, 1024, 2048);
    gemm_bf16_wmma<<<dim3(1024/64, BROWS/128), 256, 0, stream>>>(
        inA, w1fc, fc1b, x, nullptr, BROWS, 1024, 2048, 1);

    // --- heads: shape params / glob / cam ---
    heads_kernel<<<BROWS, 128, 0, stream>>>(x, shW, shb, gW, gb, cW, cb,
                                            shp, outMu, outStd, outGl, outCam);

    // --- embed = elu(concat(input, shape, glob, cam) @ We + be) ---
    pack_embed_kernel<<<gblk((size_t)BROWS*2080), 256, 0, stream>>>(in_feats, shp, outGl, outCam, embA);
    pack_w_bf16<<<gblk((size_t)2080*512), 256, 0, stream>>>(eW, wEmb, 2077, 512, 2080);
    gemm_bf16_wmma<<<dim3(512/64, BROWS/128), 256, 0, stream>>>(
        embA, wEmb, eb, nullptr, emb, BROWS, 512, 2080, 1);

    // --- sequential kinematic chain over 23 joints ---
    for (int j = 0; j < NJ; ++j) {
        int Kp = kpad[j];
        pack_w_bf16<<<gblk((size_t)Kp*256), 256, 0, stream>>>(l1W[j], w1p[j], indim[j], 256, Kp);
        Par8 par = {};
        for (int i = 0; i < kNP[j]; ++i) par.p[i] = kP[j][i];
        pack_feat_kernel<<<gblk((size_t)BROWS*Kp), 256, 0, stream>>>(
            emb, UpA, SpA, outR, par, kNP[j], Kp, feat);
        gemm_bf16_wmma<<<dim3(256/64, BROWS/128), 256, 0, stream>>>(
            feat, w1p[j], l1b[j], hbuf, nullptr, BROWS, 256, Kp, 1);
        l2_svd_kernel<<<BROWS/128, 128, 0, stream>>>(
            hbuf, l2W[j], l2b[j], j, outF, outU, outS, outV, outR, UpA, SpA);
    }
}
